// GIN_76484777607283
// MI455X (gfx1250) — matrix operand order repaired and checked by reading, not compiled, not run on hardware
//
#include <hip/hip_runtime.h>
#include <hip/hip_bf16.h>

typedef __attribute__((ext_vector_type(16))) _Float16 v16h;
typedef __attribute__((ext_vector_type(8)))  _Float16 v8h;
typedef __attribute__((ext_vector_type(4)))  _Float16 v4h;
typedef __attribute__((ext_vector_type(8)))  float    v8f;

// ---------------------------------------------------------------------------
// Phase 1: edge-weight normalization (deg = segment_sum(ew, dst); wn = ew/deg[dst])
// ---------------------------------------------------------------------------
__global__ void k_deg(const float* __restrict__ ew, const int* __restrict__ dst,
                      float* __restrict__ deg, int E) {
    int e = blockIdx.x * blockDim.x + threadIdx.x;
    if (e < E) atomicAdd(&deg[dst[e]], ew[e]);
}

__global__ void k_wnorm(const float* __restrict__ ew, const int* __restrict__ dst,
                        const float* __restrict__ deg, float* __restrict__ wn, int E) {
    int e = blockIdx.x * blockDim.x + threadIdx.x;
    if (e < E) wn[e] = ew[e] / deg[dst[e]];
}

// ---------------------------------------------------------------------------
// Phase 2: scatter w_norm * x[src] -> agg[dst]   (F=64 f32 features)
// Half-wave per edge: 16 lanes x float4 = full 256B row — coalesced gather+scatter.
// ---------------------------------------------------------------------------
__global__ void k_scatter_f32(const float* __restrict__ x, const float* __restrict__ wn,
                              const int* __restrict__ src, const int* __restrict__ dst,
                              float* __restrict__ agg, int E) {
    int tid = blockIdx.x * blockDim.x + threadIdx.x;
    int c = tid & 15;          // 16 chunks of float4 == 64 feats
    int e = tid >> 4;
    if (e >= E) return;
    float w = wn[e];
    const float4 v = *(const float4*)(x + (size_t)src[e] * 64 + c * 4);
    float* o = agg + (size_t)dst[e] * 64 + c * 4;
    atomicAdd(o + 0, w * v.x);
    atomicAdd(o + 1, w * v.y);
    atomicAdd(o + 2, w * v.z);
    atomicAdd(o + 3, w * v.w);
}

// Scatter for F=128 f16 features: one wave per edge (32 lanes x 4 halves = 256B row).
__global__ void k_scatter_f16(const _Float16* __restrict__ h, const float* __restrict__ wn,
                              const int* __restrict__ src, const int* __restrict__ dst,
                              float* __restrict__ agg, int E) {
    int tid = blockIdx.x * blockDim.x + threadIdx.x;
    int c = tid & 31;          // 32 chunks of 4 halves == 128 feats
    int e = tid >> 5;
    if (e >= E) return;
    float w = wn[e];
    const v4h v = *(const v4h*)(h + (size_t)src[e] * 128 + c * 4);
    float* o = agg + (size_t)dst[e] * 128 + c * 4;
    atomicAdd(o + 0, w * (float)v[0]);
    atomicAdd(o + 1, w * (float)v[1]);
    atomicAdd(o + 2, w * (float)v[2]);
    atomicAdd(o + 3, w * (float)v[3]);
}

// ---------------------------------------------------------------------------
// Elementwise helpers (f32+f32 -> f16, f16+f32 -> f16, f32 -> f16)
// ---------------------------------------------------------------------------
__global__ void k_add_cvt(const float* __restrict__ a, const float* __restrict__ b,
                          _Float16* __restrict__ o, int n) {
    int i = blockIdx.x * blockDim.x + threadIdx.x;
    if (i < n) o[i] = (_Float16)(a[i] + b[i]);
}
__global__ void k_addh_cvt(const _Float16* __restrict__ a, const float* __restrict__ b,
                           _Float16* __restrict__ o, int n) {
    int i = blockIdx.x * blockDim.x + threadIdx.x;
    if (i < n) o[i] = (_Float16)((float)a[i] + b[i]);
}
__global__ void k_cvt(const float* __restrict__ a, _Float16* __restrict__ o, int n) {
    int i = blockIdx.x * blockDim.x + threadIdx.x;
    if (i < n) o[i] = (_Float16)a[i];
}

__global__ void k_cnt(const int* __restrict__ gid, float* __restrict__ cnt, int n) {
    int i = blockIdx.x * blockDim.x + threadIdx.x;
    if (i < n) atomicAdd(&cnt[gid[i]], 1.0f);
}

// ---------------------------------------------------------------------------
// WMMA GEMM, layer 1:  H[M,128] = relu(A[M,64] @ W[128,64]^T + bias)
// Block = 256 threads = 8 waves; wave w owns N-tile w; grid.x = ceil(M/16).
// Fragment layouts per CDNA5 ISA 7.12.2 (wave32).
// ---------------------------------------------------------------------------
__global__ void k_gemm1_wmma(const _Float16* __restrict__ A,   // M x 64
                             const _Float16* __restrict__ Wt,  // 128 x 64 (row-major = B^T)
                             const float* __restrict__ bias,   // 128
                             _Float16* __restrict__ H,         // M x 128
                             int M) {
    const int lane = threadIdx.x & 31;
    const int wid  = threadIdx.x >> 5;   // N tile 0..7
    const int half = lane >> 4;          // 0/1 lane-half
    const int l16  = lane & 15;
    const int bm   = blockIdx.x;

    int rowA = bm * 16 + l16;  if (rowA >= M) rowA = M - 1;   // clamp (M%16==0 in practice)
    const int colB = wid * 16 + l16;

    v8f acc = {};
#pragma unroll
    for (int kt = 0; kt < 64; kt += 32) {
        // A fragment: lane holds row rowA; lows K=kt+half*8..+7, highs K=kt+16+half*8..+7
        const _Float16* ap = A + (size_t)rowA * 64 + kt + half * 8;
        v8h a_lo = *(const v8h*)(ap);
        v8h a_hi = *(const v8h*)(ap + 16);
        v16h a = __builtin_shufflevector(a_lo, a_hi,
                                         0, 1, 2, 3, 4, 5, 6, 7,
                                         8, 9, 10, 11, 12, 13, 14, 15);
        // B fragment: lane holds column colB; lows K=kt+half*8..+7, highs K=kt+16+half*8..+7
        const _Float16* bp = Wt + (size_t)colB * 64 + kt + half * 8;
        v8h b_lo = *(const v8h*)(bp);
        v8h b_hi = *(const v8h*)(bp + 16);
        v16h b = __builtin_shufflevector(b_lo, b_hi,
                                         0, 1, 2, 3, 4, 5, 6, 7,
                                         8, 9, 10, 11, 12, 13, 14, 15);
        acc = __builtin_amdgcn_wmma_f32_16x16x32_f16(false, a, false, b,
                                                     (short)0, acc, false, false);
    }
    const float bv = bias[colB];
    const int rbase = bm * 16 + half * 8;   // C/D layout: m = (lane>>4)*8 + i
    _Float16* hp = H + (size_t)rbase * 128 + colB;
    if (rbase + 7 < M) {                     // wave-uniform: full tile (always, M%16==0)
#pragma unroll
        for (int i = 0; i < 8; ++i) {
            float v = acc[i] + bv;
            v = v > 0.f ? v : 0.f;
            hp[(size_t)i * 128] = (_Float16)v;
        }
    } else {
#pragma unroll
        for (int i = 0; i < 8; ++i) {
            if (rbase + i < M) {
                float v = acc[i] + bv;
                v = v > 0.f ? v : 0.f;
                hp[(size_t)i * 128] = (_Float16)v;
            }
        }
    }
}

// ---------------------------------------------------------------------------
// WMMA GEMM, layer 2 + fused lw-weighted graph readout:
//   h2 = relu(A[M,128] @ W[128,128]^T + b2);  wh[g, c] += lw[m] * h2[m, c]
// ---------------------------------------------------------------------------
__global__ void k_gemm2_wmma_readout(const _Float16* __restrict__ A,   // M x 128
                                     const _Float16* __restrict__ Wt,  // 128 x 128
                                     const float* __restrict__ bias,   // 128
                                     const float* __restrict__ lw,     // M
                                     const int* __restrict__ gid,      // M
                                     float* __restrict__ wh,           // 64 x 128
                                     int M) {
    const int lane = threadIdx.x & 31;
    const int wid  = threadIdx.x >> 5;
    const int half = lane >> 4;
    const int l16  = lane & 15;
    const int bm   = blockIdx.x;

    int rowA = bm * 16 + l16;  if (rowA >= M) rowA = M - 1;
    const int colB = wid * 16 + l16;

    v8f acc = {};
#pragma unroll
    for (int kt = 0; kt < 128; kt += 32) {
        const _Float16* ap = A + (size_t)rowA * 128 + kt + half * 8;
        v8h a_lo = *(const v8h*)(ap);
        v8h a_hi = *(const v8h*)(ap + 16);
        v16h a = __builtin_shufflevector(a_lo, a_hi,
                                         0, 1, 2, 3, 4, 5, 6, 7,
                                         8, 9, 10, 11, 12, 13, 14, 15);
        const _Float16* bp = Wt + (size_t)colB * 128 + kt + half * 8;
        v8h b_lo = *(const v8h*)(bp);
        v8h b_hi = *(const v8h*)(bp + 16);
        v16h b = __builtin_shufflevector(b_lo, b_hi,
                                         0, 1, 2, 3, 4, 5, 6, 7,
                                         8, 9, 10, 11, 12, 13, 14, 15);
        acc = __builtin_amdgcn_wmma_f32_16x16x32_f16(false, a, false, b,
                                                     (short)0, acc, false, false);
    }
    const float bv = bias[colB];
    const int rbase = bm * 16 + half * 8;
    if (rbase + 7 < M) {                     // wave-uniform full-tile path
#pragma unroll
        for (int i = 0; i < 8; ++i) {
            int r = rbase + i;
            float v = acc[i] + bv;
            v = v > 0.f ? v : 0.f;
            atomicAdd(&wh[gid[r] * 128 + colB], v * lw[r]);
        }
    } else {
#pragma unroll
        for (int i = 0; i < 8; ++i) {
            int r = rbase + i;
            if (r < M) {
                float v = acc[i] + bv;
                v = v > 0.f ? v : 0.f;
                atomicAdd(&wh[gid[r] * 128 + colB], v * lw[r]);
            }
        }
    }
}

// ---------------------------------------------------------------------------
// Head: hg = wh / max(cnt,1);  out1 = relu(hg @ Wd^T + bd);  out = out1 @ Wc^T + bc
// Single block, f32 throughout (keeps final output accurate). 48 KB LDS.
// ---------------------------------------------------------------------------
__global__ void k_head(const float* __restrict__ wh, const float* __restrict__ cnt,
                       const float* __restrict__ Wd, const float* __restrict__ bd,
                       const float* __restrict__ Wc, const float* __restrict__ bc,
                       float* __restrict__ out) {
    __shared__ float hg[64 * 128];
    __shared__ float o1[64 * 64];
    const int t = threadIdx.x;
    for (int i = t; i < 64 * 128; i += blockDim.x) {
        int g = i >> 7;
        float c = cnt[g];
        c = c > 1.f ? c : 1.f;
        hg[i] = wh[i] / c;
    }
    __syncthreads();
    for (int i = t; i < 64 * 64; i += blockDim.x) {
        int g = i >> 6, o = i & 63;
        float s = bd[o];
        const float* row = hg + g * 128;
        const float* w   = Wd + o * 128;
#pragma unroll 8
        for (int k = 0; k < 128; ++k) s += row[k] * w[k];
        o1[i] = s > 0.f ? s : 0.f;
    }
    __syncthreads();
    for (int i = t; i < 64 * 2; i += blockDim.x) {
        int g = i >> 1, c = i & 1;
        float s = bc[c];
        const float* row = o1 + g * 64;
        const float* w   = Wc + c * 64;
#pragma unroll 8
        for (int k = 0; k < 64; ++k) s += row[k] * w[k];
        out[i] = s;   // row-major [64,2]
    }
}

// ---------------------------------------------------------------------------
extern "C" void kernel_launch(void* const* d_in, const int* in_sizes, int n_in,
                              void* d_out, int out_size, void* d_ws, size_t ws_size,
                              hipStream_t stream) {
    const float* x   = (const float*)d_in[0];   // [N,64]
    const float* ew  = (const float*)d_in[1];   // [E]
    const float* lw  = (const float*)d_in[2];   // [N]
    const int*   src = (const int*)  d_in[3];   // [E]
    const int*   dst = (const int*)  d_in[4];   // [E]
    const int*   gid = (const int*)  d_in[5];   // [N]
    const float* W1  = (const float*)d_in[6];   // [128,64]
    const float* b1  = (const float*)d_in[7];   // [128]
    const float* W2  = (const float*)d_in[8];   // [128,128]
    const float* b2  = (const float*)d_in[9];   // [128]
    const float* Wd  = (const float*)d_in[10];  // [64,128]
    const float* bd  = (const float*)d_in[11];  // [64]
    const float* Wc  = (const float*)d_in[12];  // [2,64]
    const float* bc  = (const float*)d_in[13];  // [2]

    const int N = in_sizes[0] / 64;   // 100000
    const int E = in_sizes[1];        // 1000000

    // ---- workspace carve-up (256B aligned); agg1 region reused for A2 f16 ----
    char* ws = (char*)d_ws;
    size_t off = 0;
    auto take = [&](size_t bytes) -> char* {
        char* p = ws + off;
        off = (off + bytes + 255) & ~(size_t)255;
        return p;
    };
    float*    deg  = (float*)take((size_t)N * 4);
    float*    wn   = (float*)take((size_t)E * 4);
    char*     reg  = take((size_t)N * 64 * 4);          // agg1 (f32, N*64) == A2 (f16, N*128)
    float*    agg1 = (float*)reg;
    _Float16* A2   = (_Float16*)reg;
    _Float16* A1   = (_Float16*)take((size_t)N * 64 * 2);
    _Float16* H1v  = (_Float16*)take((size_t)N * 128 * 2);
    float*    agg2 = (float*)take((size_t)N * 128 * 4);
    _Float16* W1h  = (_Float16*)take(128 * 64 * 2);
    _Float16* W2h  = (_Float16*)take(128 * 128 * 2);
    float*    wh   = (float*)take(64 * 128 * 4);
    float*    cnt  = (float*)take(64 * 4);

    const int TB = 256;

    // zero accumulators every launch (ws is poisoned, never re-cleared by harness)
    hipMemsetAsync(deg,  0, (size_t)N * 4, stream);
    hipMemsetAsync(agg1, 0, (size_t)N * 64 * 4, stream);
    hipMemsetAsync(agg2, 0, (size_t)N * 128 * 4, stream);
    hipMemsetAsync(wh,   0, 64 * 128 * 4, stream);
    hipMemsetAsync(cnt,  0, 64 * 4, stream);

    // edge-weight normalization
    k_deg  <<<(E + TB - 1) / TB, TB, 0, stream>>>(ew, dst, deg, E);
    k_wnorm<<<(E + TB - 1) / TB, TB, 0, stream>>>(ew, dst, deg, wn, E);

    // conv1 aggregate + A1 = f16(x + agg1)
    k_scatter_f32<<<(E * 16 + TB - 1) / TB, TB, 0, stream>>>(x, wn, src, dst, agg1, E);
    k_add_cvt<<<((N * 64) + TB - 1) / TB, TB, 0, stream>>>(x, agg1, A1, N * 64);

    // f16 weights
    k_cvt<<<(128 * 64  + TB - 1) / TB, TB, 0, stream>>>(W1, W1h, 128 * 64);
    k_cvt<<<(128 * 128 + TB - 1) / TB, TB, 0, stream>>>(W2, W2h, 128 * 128);

    // layer-1 GEMM (WMMA), h1 = relu(A1 @ W1^T + b1)
    k_gemm1_wmma<<<(N + 15) / 16, 256, 0, stream>>>(A1, W1h, b1, H1v, N);

    // conv2 aggregate + A2 = f16(h1 + agg2)
    k_scatter_f16<<<(E * 32 + TB - 1) / TB, TB, 0, stream>>>(H1v, wn, src, dst, agg2, E);
    k_addh_cvt<<<((N * 128) + TB - 1) / TB, TB, 0, stream>>>(H1v, agg2, A2, N * 128);

    // per-graph node counts
    k_cnt<<<(N + TB - 1) / TB, TB, 0, stream>>>(gid, cnt, N);

    // layer-2 GEMM (WMMA) with fused lw-weighted readout into wh[64,128]
    k_gemm2_wmma_readout<<<(N + 15) / 16, 256, 0, stream>>>(A2, W2h, b2, lw, gid, wh, N);

    // MLP head -> d_out [64,2] f32
    k_head<<<1, 256, 0, stream>>>(wh, cnt, Wd, bd, Wc, bc, (float*)d_out);
}
